// TransformerBlock_58798102282715
// MI455X (gfx1250) — compile-verified, matrix operand order checked by reading, not run on hardware
//
#include <hip/hip_runtime.h>
#include <hip/hip_bf16.h>

// ---------------------------------------------------------------------------
// MI455X (gfx1250) transformer block.
// Compute-bound (~172 GFLOP vs <100MB unique bytes; weights live in 192MB L2)
// => all matrix math on v_wmma_f32_16x16x32_bf16 (wave32), bf16 operands,
//    f32 accumulation.  GEMM A-tiles double-buffered in LDS via the CDNA5
//    async path (global_load_async_to_lds_b128 + s_wait_asynccnt).  Softmax
//    row-sums via WMMA-with-ones, row-max via DPP16 butterflies, exponentials
//    in the base-2 domain (q pre-scaled by 0.125*log2e -> raw v_exp_f32).
// ---------------------------------------------------------------------------

typedef __bf16 bf16;
typedef __attribute__((ext_vector_type(16))) __bf16 v16bf;
typedef __attribute__((ext_vector_type(8)))  __bf16 v8bf;
typedef __attribute__((ext_vector_type(8)))  float  v8f;

// Async copy of 16 bytes/lane from global (SADDR base + 32-bit VGPR byte
// offset, GVS mode) into LDS (VGPR holds LDS byte address = low 32 bits of
// the generic pointer).  Tracked by ASYNCcnt (ISA 10.7 / 15.18.3).
#define ASYNC_CP_B128(ldsaddr, base, voff)                                    \
  asm volatile("global_load_async_to_lds_b128 %0, %1, %2"                     \
               :: "v"(ldsaddr), "v"(voff), "s"(base) : "memory")
#define WAIT_ASYNC(n) asm volatile("s_wait_asynccnt " #n ::: "memory")

// ----- CDNA5 16-bit A/B fragment loader ------------------------------------
// 16-bit A-matrix 16x32 layout (ISA 7.12.2): lanes 0-15 hold row M=lane,
// K=0..7 in v0..3 and K=16..23 in v4..7; lanes 16-31 same rows, K=8..15/24..31.
// B fragments are loaded with the identical pattern from an (N x K) row-major
// matrix (K contiguous per output column): weights / K / V^T are stored so.
static __device__ __forceinline__ v16bf load_frag(const bf16* base, int ld,
                                                  int row, int k0, int lane) {
  int r  = row + (lane & 15);
  int kg = (lane >> 4) & 1;              // lanes 16-31 take the +8 K group
  const bf16* p = base + (size_t)r * ld + k0 + kg * 8;
  v8bf lo = *reinterpret_cast<const v8bf*>(p);
  v8bf hi = *reinterpret_cast<const v8bf*>(p + 16);
  v16bf v;
#pragma unroll
  for (int i = 0; i < 8; ++i) { v[i] = lo[i]; v[i + 8] = hi[i]; }
  return v;
}

static __device__ __forceinline__ v8f wmma_bf16f32(v16bf a, v16bf b, v8f c) {
  return __builtin_amdgcn_wmma_f32_16x16x32_bf16(false, a, false, b,
                                                 (short)0, c, false, false);
}

static __device__ __forceinline__ v16bf ones_frag() {
  v16bf v;
#pragma unroll
  for (int i = 0; i < 16; ++i) v[i] = (bf16)1.0f;
  return v;
}

// ----- 16-lane row max via DPP16 butterflies (VALU only, no DS pipe) --------
// xor1 = quad_perm(1,0,3,2)=0xB1, xor2 = quad_perm(2,3,0,1)=0x4E,
// xor7 = row_half_mirror=0x141, xor15 = row_mirror=0x140.
template <int CTRL>
static __device__ __forceinline__ float fmax_dpp(float x) {
  int v = __builtin_amdgcn_update_dpp(0, __float_as_int(x), CTRL, 0xF, 0xF, true);
  return fmaxf(x, __int_as_float(v));
}
static __device__ __forceinline__ float rowmax16(float x) {
  x = fmax_dpp<0xB1>(x);
  x = fmax_dpp<0x4E>(x);
  x = fmax_dpp<0x141>(x);
  x = fmax_dpp<0x140>(x);
  return x;
}

// ----- fp32 -> bf16 weight conversion ---------------------------------------
__global__ void tb_cvt_f32_bf16(const float* __restrict__ in,
                                bf16* __restrict__ out, int n) {
  int i = blockIdx.x * 256 + threadIdx.x;
  if (i < n) out[i] = (bf16)in[i];
}

// ----- fused RMSNorm + bf16 cast --------------------------------------------
__global__ __launch_bounds__(256) void tb_rmsnorm_bf16(
    const float* __restrict__ x, const float* __restrict__ w,
    bf16* __restrict__ out, int D) {
  int row = blockIdx.x;
  const float* xr = x + (size_t)row * D;
  float ss = 0.f;
  for (int i = threadIdx.x; i < D; i += 256) { float v = xr[i]; ss += v * v; }
#pragma unroll
  for (int off = 16; off > 0; off >>= 1) ss += __shfl_xor(ss, off, 32);
  __shared__ float red[8];
  if ((threadIdx.x & 31) == 0) red[threadIdx.x >> 5] = ss;
  __syncthreads();
  float tot = 0.f;
#pragma unroll
  for (int i = 0; i < 8; ++i) tot += red[i];
  float scale = rsqrtf(tot / (float)D + 1e-6f);
  for (int i = threadIdx.x; i < D; i += 256)
    out[(size_t)row * D + i] = (bf16)(xr[i] * scale * w[i]);
}

// ----- bf16 WMMA GEMM: C[M,N] = A[M,K] @ W[N,K]^T (+ epilogue) --------------
// 256 threads = 8 waves as 2x4; wave tile 32x64 (2x4 accums); block 64x256.
// A staged as 64x64 bf16 tiles (8KB), double-buffered via async-to-LDS DMA
// (2 x b128 per thread per tile); 16 wmma between barrier pairs.
// EPI: 0 = f32 store, 1 = f32 residual add, 2 = bf16 store.
template <int EPI>
__global__ __launch_bounds__(256) void tb_gemm_bf16(
    const bf16* __restrict__ A, const bf16* __restrict__ W,
    void* __restrict__ Cv, const float* __restrict__ R,
    int M, int N, int K) {
  __shared__ __attribute__((aligned(16))) bf16 aBuf[2][64][64];

  int lane   = threadIdx.x & 31;
  int wave   = threadIdx.x >> 5;
  int mBlock = blockIdx.y * 64;
  int mLoc   = (wave >> 2) * 32;               // wave's rows inside the block
  int nBase  = blockIdx.x * 256 + (wave & 3) * 64;

  // per-thread async-copy slot: 32B = 16 bf16 of one A row (2 x b128)
  int cpRow = threadIdx.x >> 2;                // 0..63
  int cpCol = (threadIdx.x & 3) * 16;          // 0,16,32,48
  unsigned ldsA[2] = {
      (unsigned)(uintptr_t)&aBuf[0][cpRow][cpCol],
      (unsigned)(uintptr_t)&aBuf[1][cpRow][cpCol]};
  unsigned gOff = (unsigned)((mBlock + cpRow) * K + cpCol) * 2u;

  int Kt = K >> 6;                             // 64-wide K tiles
  ASYNC_CP_B128(ldsA[0],      A, gOff);
  ASYNC_CP_B128(ldsA[0] + 16, A, gOff + 16);
  if (Kt > 1) {
    ASYNC_CP_B128(ldsA[1],      A, gOff + 128u);
    ASYNC_CP_B128(ldsA[1] + 16, A, gOff + 144u);
  }

  v8f acc[2][4] = {};
  for (int kt = 0; kt < Kt; ++kt) {
    // async loads complete in order: <=2 outstanding means tile kt is done
    if (kt + 1 < Kt) { WAIT_ASYNC(0x2); } else { WAIT_ASYNC(0x0); }
    __syncthreads();                           // tile kt visible to all waves

    if (kt + 1 < Kt) {                         // prefetch next B tile into L2/L0
      __builtin_prefetch(W + (size_t)(nBase + lane) * K + (kt + 1) * 64, 0, 1);
      __builtin_prefetch(W + (size_t)(nBase + 32 + lane) * K + (kt + 1) * 64, 0, 1);
    }

    const bf16* ab = &aBuf[kt & 1][0][0];
#pragma unroll
    for (int kh = 0; kh < 2; ++kh) {
      int k0 = (kt << 6) + (kh << 5);
      v16bf a0 = load_frag(ab, 64, mLoc,      kh << 5, lane);   // ds_load
      v16bf a1 = load_frag(ab, 64, mLoc + 16, kh << 5, lane);
      v16bf b0 = load_frag(W, K, nBase,      k0, lane);
      v16bf b1 = load_frag(W, K, nBase + 16, k0, lane);
      v16bf b2 = load_frag(W, K, nBase + 32, k0, lane);
      v16bf b3 = load_frag(W, K, nBase + 48, k0, lane);
      acc[0][0] = wmma_bf16f32(a0, b0, acc[0][0]);
      acc[0][1] = wmma_bf16f32(a0, b1, acc[0][1]);
      acc[0][2] = wmma_bf16f32(a0, b2, acc[0][2]);
      acc[0][3] = wmma_bf16f32(a0, b3, acc[0][3]);
      acc[1][0] = wmma_bf16f32(a1, b0, acc[1][0]);
      acc[1][1] = wmma_bf16f32(a1, b1, acc[1][1]);
      acc[1][2] = wmma_bf16f32(a1, b2, acc[1][2]);
      acc[1][3] = wmma_bf16f32(a1, b3, acc[1][3]);
    }

    __syncthreads();                           // all reads of tile kt done
    if (kt + 2 < Kt) {                         // refill the buffer just freed
      unsigned go = gOff + (unsigned)(kt + 2) * 128u;
      ASYNC_CP_B128(ldsA[kt & 1],      A, go);
      ASYNC_CP_B128(ldsA[kt & 1] + 16, A, go + 16u);
    }
  }

  // C/D layout: lane l owns column (l&15); VGPR i -> row i + 8*(l>>4).
  int col0   = nBase + (lane & 15);
  int rowOff = (lane >> 4) * 8;
#pragma unroll
  for (int ti = 0; ti < 2; ++ti)
#pragma unroll
    for (int tj = 0; tj < 4; ++tj)
#pragma unroll
      for (int i = 0; i < 8; ++i) {
        int r = mBlock + mLoc + ti * 16 + rowOff + i;
        int c = col0 + tj * 16;
        size_t idx = (size_t)r * N + c;
        float v = acc[ti][tj][i];
        if (EPI == 0)      ((float*)Cv)[idx] = v;
        else if (EPI == 1) ((float*)Cv)[idx] = R[idx] + v;
        else               ((bf16*)Cv)[idx] = (bf16)v;
      }
}

// ----- RoPE + head split + V transpose --------------------------------------
// q is pre-scaled by 0.125*log2(e) so attention scores come out of the WMMA
// already in the base-2 softmax domain (no per-element scaling, raw v_exp).
__global__ void tb_rope_qkv(const float* __restrict__ qkv,
                            const int* __restrict__ sp,
                            bf16* __restrict__ qr, bf16* __restrict__ kr,
                            bf16* __restrict__ vt) {
  const float QS = 0.125f * 1.44269504088896340736f;  // (1/sqrt(64))*log2(e)
  int tid = blockIdx.x * 256 + threadIdx.x;   // [0, 32*2048*32)
  int i  = tid & 31;                          // dim pair
  int s  = (tid >> 5) & 2047;                 // seq pos
  int bh = tid >> 16;                         // 0..31
  int b = bh >> 4, h = bh & 15;
  size_t qb = ((size_t)(b * 2048 + s)) * 3072 + h * 64;
  float qe = qkv[qb + 2 * i],        qo = qkv[qb + 2 * i + 1];
  float ke = qkv[qb + 1024 + 2 * i], ko = qkv[qb + 1024 + 2 * i + 1];
  float v0 = qkv[qb + 2048 + 2 * i], v1 = qkv[qb + 2048 + 2 * i + 1];
  float freq = __expf(-(float)(2 * i) * (9.2103403719761827f / 64.0f));
  float t = (float)(sp[0] + s);
  float sn, cs;
  __sincosf(t * freq, &sn, &cs);
  size_t ob = ((size_t)bh * 2048 + s) * 64;
  qr[ob + 2 * i]     = (bf16)((qe * cs - qo * sn) * QS);
  qr[ob + 2 * i + 1] = (bf16)((qe * sn + qo * cs) * QS);
  kr[ob + 2 * i]     = (bf16)(ke * cs - ko * sn);
  kr[ob + 2 * i + 1] = (bf16)(ke * sn + ko * cs);
  size_t vb = (size_t)bh * 64 * 2048;
  vt[vb + (size_t)(2 * i) * 2048 + s]     = (bf16)v0;
  vt[vb + (size_t)(2 * i + 1) * 2048 + s] = (bf16)v1;
}

// ----- Flash attention: one 32-key tile step --------------------------------
// MASKED=false for tiles fully below the diagonal (no v_cmp/v_cndmask).
// Row-max: DPP16 butterflies.  Row-sum: P @ ones on the WMMA pipe.
// All exponentials are exp2 (scores already in base-2 domain).
template <bool MASKED>
static __device__ __forceinline__ void attn_tile(
    int k0, v16bf aq0, v16bf aq1, const bf16* kh, const bf16* vh,
    bf16 (*pT)[32], v8f* o, float* mrow, float* lrow,
    int qRow0, int rowOff, int nlane, int lane) {
  const int S = 2048, HD = 64;
  v8f s0 = {}, s1 = {};
  s0 = wmma_bf16f32(aq0, load_frag(kh, HD, k0,      0,  lane), s0);
  s0 = wmma_bf16f32(aq1, load_frag(kh, HD, k0,      32, lane), s0);
  s1 = wmma_bf16f32(aq0, load_frag(kh, HD, k0 + 16, 0,  lane), s1);
  s1 = wmma_bf16f32(aq1, load_frag(kh, HD, k0 + 16, 32, lane), s1);

  float alpha[8];
#pragma unroll
  for (int i = 0; i < 8; ++i) {
    float v0 = s0[i];
    float v1 = s1[i];
    if (MASKED) {
      int row = qRow0 + rowOff + i;
      if (k0 + nlane > row)      v0 = -3.0e38f;
      if (k0 + 16 + nlane > row) v1 = -3.0e38f;
    }
    float mx = rowmax16(fmaxf(v0, v1));        // 16-lane row-group max (DPP)
    float nm = fmaxf(mrow[i], mx);
    float a  = __builtin_amdgcn_exp2f(mrow[i] - nm);   // raw v_exp_f32
    float p0 = __builtin_amdgcn_exp2f(v0 - nm);
    float p1 = __builtin_amdgcn_exp2f(v1 - nm);
    alpha[i] = a;
    mrow[i]  = nm;
    pT[rowOff + i][nlane]      = (bf16)p0;     // C-layout -> LDS (A-layout src)
    pT[rowOff + i][nlane + 16] = (bf16)p1;
  }
#pragma unroll
  for (int c = 0; c < 4; ++c)
#pragma unroll
    for (int i = 0; i < 8; ++i) o[c][i] *= alpha[i];

  v16bf pf = load_frag(&pT[0][0], 32, 0, 0, lane);   // ds_load, same-wave order
  v8f sums = {};
  sums = wmma_bf16f32(pf, ones_frag(), sums);  // row sums, replicated per lane
#pragma unroll
  for (int i = 0; i < 8; ++i) lrow[i] = lrow[i] * alpha[i] + sums[i];

#pragma unroll
  for (int c = 0; c < 4; ++c)      // P(16x32) @ V(32x64), hd in 4 chunks of 16
    o[c] = wmma_bf16f32(pf, load_frag(vh, S, c * 16, k0, lane), o[c]);
}

// grid (S/64, B*H), 128 threads = 4 waves, one wave per 16 query rows.
__global__ __launch_bounds__(128) void tb_flash_attn(
    const bf16* __restrict__ qr, const bf16* __restrict__ kr,
    const bf16* __restrict__ vt, bf16* __restrict__ attn) {
  const int S = 2048, HD = 64;
  __shared__ __attribute__((aligned(16))) bf16 pTile[4][16][32];

  int lane = threadIdx.x & 31;
  int wave = threadIdx.x >> 5;
  int bh   = blockIdx.y;
  int qRow0 = blockIdx.x * 64 + wave * 16;
  const bf16* qh = qr + (size_t)bh * S * HD;
  const bf16* kh = kr + (size_t)bh * S * HD;
  const bf16* vh = vt + (size_t)bh * HD * S;

  v16bf aq0 = load_frag(qh, HD, qRow0, 0, lane);
  v16bf aq1 = load_frag(qh, HD, qRow0, 32, lane);

  v8f o[4] = {};
  float mrow[8], lrow[8];
#pragma unroll
  for (int i = 0; i < 8; ++i) { mrow[i] = -1e30f; lrow[i] = 0.f; }
  int rowOff = (lane >> 4) * 8;
  int nlane  = lane & 15;

  int nFull = qRow0 >> 5;                      // tiles with k0+32 <= qRow0
  for (int kt = 0; kt < nFull; ++kt)
    attn_tile<false>(kt << 5, aq0, aq1, kh, vh, pTile[wave], o, mrow, lrow,
                     qRow0, rowOff, nlane, lane);
  attn_tile<true>(nFull << 5, aq0, aq1, kh, vh, pTile[wave], o, mrow, lrow,
                  qRow0, rowOff, nlane, lane); // single diagonal tile

  int b = bh >> 4, h = bh & 15;
#pragma unroll
  for (int c = 0; c < 4; ++c)
#pragma unroll
    for (int i = 0; i < 8; ++i) {
      int q = qRow0 + rowOff + i;
      float v = o[c][i] / lrow[i];
      size_t idx = ((size_t)(b * 2048 + q)) * 1024 + h * 64 + c * 16 + nlane;
      attn[idx] = (bf16)v;
    }
}

// ----- SiLU(gate) * up ------------------------------------------------------
__global__ void tb_silu_mul(const bf16* __restrict__ g,
                            const bf16* __restrict__ u,
                            bf16* __restrict__ h, int n) {
  int i = blockIdx.x * 256 + threadIdx.x;
  if (i < n) {
    float gv = (float)g[i], uv = (float)u[i];
    float s = gv / (1.f + __expf(-gv));
    h[i] = (bf16)(s * uv);
  }
}

// ---------------------------------------------------------------------------
extern "C" void kernel_launch(void* const* d_in, const int* in_sizes, int n_in,
                              void* d_out, int out_size, void* d_ws, size_t ws_size,
                              hipStream_t stream) {
  (void)in_sizes; (void)n_in; (void)out_size; (void)ws_size;
  const float* x    = (const float*)d_in[0];
  const int*   sp   = (const int*)d_in[1];
  const float* n1w  = (const float*)d_in[2];
  const float* wqkv = (const float*)d_in[3];
  const float* wo   = (const float*)d_in[4];
  const float* n2w  = (const float*)d_in[5];
  const float* wg   = (const float*)d_in[6];
  const float* wu   = (const float*)d_in[7];
  const float* wd   = (const float*)d_in[8];
  float* out = (float*)d_out;

  // Workspace layout (~240 MB, regions 16B aligned).
  const size_t T = 4096;                             // B*S tokens
  bf16*  bWq  = (bf16*)d_ws;                         // 3072*1024
  bf16*  bWo  = bWq + (size_t)3072 * 1024;           // 1024*1024
  bf16*  bWg  = bWo + (size_t)1024 * 1024;           // 4096*1024
  bf16*  bWu  = bWg + (size_t)4096 * 1024;           // 4096*1024
  bf16*  bWd  = bWu + (size_t)4096 * 1024;           // 1024*4096
  bf16*  xn   = bWd + (size_t)1024 * 4096;           // T*1024 bf16
  float* qkv  = (float*)(xn + T * 1024);             // T*3072 f32
  bf16*  qrb  = (bf16*)(qkv + T * 3072);             // 32*2048*64
  bf16*  krb  = qrb + (size_t)32 * 2048 * 64;
  bf16*  vtb  = krb + (size_t)32 * 2048 * 64;
  bf16*  attnb= vtb + (size_t)32 * 2048 * 64;        // T*1024 bf16
  float* out1 = (float*)(attnb + T * 1024);          // T*1024 f32
  bf16*  xn2  = (bf16*)(out1 + T * 1024);            // T*1024 bf16
  bf16*  gateb= xn2 + T * 1024;                      // T*4096 bf16
  bf16*  upb  = gateb + T * 4096;
  bf16*  hb   = upb + T * 4096;

  // 1) weights -> bf16 once; every GEMM re-reads them from L2 at half width
  tb_cvt_f32_bf16<<<(3072 * 1024 + 255) / 256, 256, 0, stream>>>(wqkv, bWq, 3072 * 1024);
  tb_cvt_f32_bf16<<<(1024 * 1024 + 255) / 256, 256, 0, stream>>>(wo,   bWo, 1024 * 1024);
  tb_cvt_f32_bf16<<<(4096 * 1024 + 255) / 256, 256, 0, stream>>>(wg,   bWg, 4096 * 1024);
  tb_cvt_f32_bf16<<<(4096 * 1024 + 255) / 256, 256, 0, stream>>>(wu,   bWu, 4096 * 1024);
  tb_cvt_f32_bf16<<<(1024 * 4096 + 255) / 256, 256, 0, stream>>>(wd,   bWd, 1024 * 4096);

  // 2) xn = rmsnorm(x)
  tb_rmsnorm_bf16<<<(int)T, 256, 0, stream>>>(x, n1w, xn, 1024);

  // 3) qkv = xn @ Wqkv^T
  tb_gemm_bf16<0><<<dim3(3072 / 256, T / 64), 256, 0, stream>>>(
      xn, bWq, (void*)qkv, nullptr, (int)T, 3072, 1024);

  // 4) RoPE + head split + V^T (q pre-scaled into base-2 softmax domain)
  tb_rope_qkv<<<(32 * 2048 * 32) / 256, 256, 0, stream>>>(qkv, sp, qrb, krb, vtb);

  // 5) flash attention -> attn (bf16, [B,S,D])
  tb_flash_attn<<<dim3(2048 / 64, 32), 128, 0, stream>>>(qrb, krb, vtb, attnb);

  // 6) out1 = x + attn @ Wo^T
  tb_gemm_bf16<1><<<dim3(1024 / 256, T / 64), 256, 0, stream>>>(
      attnb, bWo, (void*)out1, x, (int)T, 1024, 1024);

  // 7) xn2 = rmsnorm(out1)
  tb_rmsnorm_bf16<<<(int)T, 256, 0, stream>>>(out1, n2w, xn2, 1024);

  // 8) gate / up projections
  tb_gemm_bf16<2><<<dim3(4096 / 256, T / 64), 256, 0, stream>>>(
      xn2, bWg, (void*)gateb, nullptr, (int)T, 4096, 1024);
  tb_gemm_bf16<2><<<dim3(4096 / 256, T / 64), 256, 0, stream>>>(
      xn2, bWu, (void*)upb, nullptr, (int)T, 4096, 1024);

  // 9) h = silu(gate) * up
  tb_silu_mul<<<(16777216) / 256, 256, 0, stream>>>(gateb, upb, hb, 16777216);

  // 10) out = out1 + h @ Wd^T
  tb_gemm_bf16<1><<<dim3(1024 / 256, T / 64), 256, 0, stream>>>(
      hb, bWd, (void*)out, out1, (int)T, 1024, 4096);
}
